// ScaledDotProductAttention_72378788872802
// MI455X (gfx1250) — matrix operand order repaired and checked by reading, not compiled, not run on hardware
//
#include <hip/hip_runtime.h>
#include <stdint.h>

#define DM    1024
#define NHEAD 16
#define DH    64
#define NQ    2048
#define NK    2048

typedef __attribute__((ext_vector_type(16))) __bf16       v16bf;
typedef __attribute__((ext_vector_type(8)))  float        v8f;
typedef __attribute__((ext_vector_type(4)))  unsigned int u32x4;
typedef __attribute__((ext_vector_type(8)))  unsigned int u32x8;
typedef int v4i __attribute__((vector_size(16)));   // matches async-LDS builtin param

#define AS1 __attribute__((address_space(1)))
#define AS3 __attribute__((address_space(3)))

// ---------- helpers ----------

__device__ __forceinline__ unsigned short f2bf(float f) {
  unsigned int u = __float_as_uint(f);
  u += 0x7FFFu + ((u >> 16) & 1u);           // round-to-nearest-even
  return (unsigned short)(u >> 16);
}

__device__ __forceinline__ v16bf pack_frag(u32x4 lo, u32x4 hi) {
  u32x8 u;
  u[0] = lo[0]; u[1] = lo[1]; u[2] = lo[2]; u[3] = lo[3];
  u[4] = hi[0]; u[5] = hi[1]; u[6] = hi[2]; u[7] = hi[3];
  return __builtin_bit_cast(v16bf, u);
}

// A-matrix fragment (16 rows x 32 k, bf16) from row-major [rows][ld]:
// lane L -> row = row0 + (L&15); k in {k0..k0+7, k0+16..k0+23}, k0 = kbase + ((L>>4)<<3)
__device__ __forceinline__ v16bf load_a_frag(const unsigned short* p, int row0,
                                             int ld, int kbase, int lane) {
  int r  = row0 + (lane & 15);
  int k0 = kbase + ((lane >> 4) << 3);
  const unsigned short* q = p + (size_t)r * ld + k0;
  u32x4 lo = *(const u32x4*)q;
  u32x4 hi = *(const u32x4*)(q + 16);
  return pack_frag(lo, hi);
}

// B-matrix fragment (32 k x 16 cols, bf16) sourced from row-major [N][K]
// (computes with B = src^T): lane L -> col = col0 + (L&15);
// k in {k0..k0+7, k0+16..k0+23}, k0 = kbase + ((L>>4)<<3)
__device__ __forceinline__ v16bf load_b_frag(const unsigned short* p, int col0,
                                             int ld, int kbase, int lane) {
  int n  = col0 + (lane & 15);
  int k0 = kbase + ((lane >> 4) << 3);
  const unsigned short* q = p + (size_t)n * ld + k0;
  u32x4 lo = *(const u32x4*)q;
  u32x4 hi = *(const u32x4*)(q + 16);
  return pack_frag(lo, hi);
}

__device__ __forceinline__ v8f wmma_bf16(v16bf a, v16bf b, v8f c) {
  return __builtin_amdgcn_wmma_f32_16x16x32_bf16(false, a, false, b,
                                                 (short)0, c, false, false);
}

// 16-byte global -> LDS copy: async DMA (ASYNCcnt) when the builtin exists,
// otherwise bounce through VGPRs (ds_store_b128).
__device__ __forceinline__ void copy16_g2l(const unsigned short* g, unsigned short* l) {
#if defined(__has_builtin) && __has_builtin(__builtin_amdgcn_global_load_async_to_lds_b128)
  AS1 v4i* gp = (AS1 v4i*)(unsigned long long)(uintptr_t)g;
  AS3 v4i* lp = (AS3 v4i*)(unsigned int)(uintptr_t)l;
  __builtin_amdgcn_global_load_async_to_lds_b128(gp, lp, 0, 0);
#else
  *(u32x4*)l = *(const u32x4*)g;
#endif
}

__device__ __forceinline__ void wait_async_le2() {
#if defined(__has_builtin) && __has_builtin(__builtin_amdgcn_s_wait_asynccnt)
  __builtin_amdgcn_s_wait_asynccnt(2);
#else
  asm volatile("s_wait_asynccnt 0x2" ::: "memory");
#endif
}

__device__ __forceinline__ void wait_async_0() {
#if defined(__has_builtin) && __has_builtin(__builtin_amdgcn_s_wait_asynccnt)
  __builtin_amdgcn_s_wait_asynccnt(0);
#else
  asm volatile("s_wait_asynccnt 0x0" ::: "memory");
#endif
}

// ---------- elementwise kernels ----------

__global__ void cvt_bf16_kernel(const float* __restrict__ in,
                                unsigned short* __restrict__ out, int n) {
  int i = blockIdx.x * blockDim.x + threadIdx.x;
  if (i < n) out[i] = f2bf(in[i]);
}

__global__ void zero_f32_kernel(float* __restrict__ p, int n) {
  int i = blockIdx.x * blockDim.x + threadIdx.x;
  if (i < n) p[i] = 0.0f;
}

// ---------- GEMM: C(MxN) = A(bf16, MxK, rm) * Bw^T (Bw bf16 NxK rm) + bias ----------
// 4 waves / block; block tile 128(M) x 64(N); wave tile 32(M) x 64(N).
// B panel (64 cols x 32 k = 4 KB) staged into double-buffered LDS with async
// global->LDS copies, shared by all 4 waves.
// mode 0: out bf16 row-major [M][N]
// mode 1: out bf16 head-transposed: out[h][d][m], h = col/64, d = col%64  (for V)
// mode 2: out f32 row-major [M][N]
__global__ void gemm_bf16_kernel(const unsigned short* __restrict__ A,
                                 const unsigned short* __restrict__ Bw,
                                 const float* __restrict__ bias,
                                 void* __restrict__ out,
                                 int M, int N, int K, int mode) {
  __shared__ __attribute__((aligned(16))) unsigned short Bp[2][64 * 32];

  const int tid  = threadIdx.x;
  const int lane = tid & 31;
  const int wave = tid >> 5;
  const int tn0  = blockIdx.x * 64;              // block col base
  const int tmw  = blockIdx.y * 128 + wave * 32; // wave row base

  // staging assignment: 256 16B-chunks per panel; thread copies idx=tid, tid+128
  const int n_s0 = tid >> 2,          c_s0 = tid & 3;
  const int n_s1 = (tid + 128) >> 2,  c_s1 = (tid + 128) & 3;
  const unsigned short* g_s0 = Bw + (size_t)(tn0 + n_s0) * K + c_s0 * 8;
  const unsigned short* g_s1 = Bw + (size_t)(tn0 + n_s1) * K + c_s1 * 8;
  unsigned short* l_s0_0 = &Bp[0][n_s0 * 32 + c_s0 * 8];
  unsigned short* l_s1_0 = &Bp[0][n_s1 * 32 + c_s1 * 8];
  unsigned short* l_s0_1 = &Bp[1][n_s0 * 32 + c_s0 * 8];
  unsigned short* l_s1_1 = &Bp[1][n_s1 * 32 + c_s1 * 8];

  v8f acc[2][4];
#pragma unroll
  for (int mi = 0; mi < 2; ++mi)
#pragma unroll
    for (int ni = 0; ni < 4; ++ni)
#pragma unroll
      for (int r = 0; r < 8; ++r) acc[mi][ni][r] = 0.0f;

  // prologue: stage chunk 0 into buffer 0
  copy16_g2l(g_s0, l_s0_0);
  copy16_g2l(g_s1, l_s1_0);

  const int nchunk = K / 32;
  for (int c = 0; c < nchunk; ++c) {
    const int k   = c * 32;
    const int cur = c & 1;

    // stage next chunk into the other buffer while this one computes
    if (c + 1 < nchunk) {
      unsigned short* l0 = cur ? l_s0_0 : l_s0_1;
      unsigned short* l1 = cur ? l_s1_0 : l_s1_1;
      copy16_g2l(g_s0 + k + 32, l0);
      copy16_g2l(g_s1 + k + 32, l1);
      wait_async_le2();   // the 2 just-issued may be in flight; chunk c is done
    } else {
      wait_async_0();
    }
    __syncthreads();      // panel for chunk c visible to all 4 waves

    if (k + 256 < K)      // global_prefetch_b8 for A stream
      __builtin_prefetch(A + (size_t)(tmw + (lane & 15)) * K + k + 256);

    v16bf a0 = load_a_frag(A, tmw,      K, k, lane);
    v16bf a1 = load_a_frag(A, tmw + 16, K, k, lane);
#pragma unroll
    for (int ni = 0; ni < 4; ++ni) {
      v16bf b = load_b_frag(&Bp[cur][0], ni * 16, 32, 0, lane);
      acc[0][ni] = wmma_bf16(a0, b, acc[0][ni]);
      acc[1][ni] = wmma_bf16(a1, b, acc[1][ni]);
    }
    __syncthreads();      // done reading buf[cur] before it is re-staged
  }

  const int rb = (lane >> 4) << 3;
#pragma unroll
  for (int mi = 0; mi < 2; ++mi) {
#pragma unroll
    for (int ni = 0; ni < 4; ++ni) {
      const int col = tn0 + ni * 16 + (lane & 15);
      const float bv = bias[col];
      const int row0 = tmw + mi * 16 + rb;
      if (mode == 0) {
        unsigned short* o = (unsigned short*)out;
#pragma unroll
        for (int r = 0; r < 8; ++r)
          o[(size_t)(row0 + r) * N + col] = f2bf(acc[mi][ni][r] + bv);
      } else if (mode == 1) {
        unsigned short* o = (unsigned short*)out;
        const int h = col >> 6, d = col & 63;
        const size_t base = ((size_t)h * DH + d) * (size_t)M;
#pragma unroll
        for (int r = 0; r < 8; ++r)
          o[base + (size_t)(row0 + r)] = f2bf(acc[mi][ni][r] + bv);
      } else {
        float* o = (float*)out;
#pragma unroll
        for (int r = 0; r < 8; ++r)
          o[(size_t)(row0 + r) * N + col] = acc[mi][ni][r] + bv;
      }
    }
  }
}

// ---------- fused flash attention per (head, 16-row q tile) ----------
// Qh, Kh: bf16 [2048][1024] row-major; Vt: bf16 [H][DH][NK] (d-major per head)
// Writes Ohb bf16 [2048][1024] (merged heads) and atomically accumulates
// pre-softmax scaled scores / NHEAD into att_map (f32 [NQ][NK]).
__global__ void flash_attn_kernel(const unsigned short* __restrict__ Qh,
                                  const unsigned short* __restrict__ Kh,
                                  const unsigned short* __restrict__ Vt,
                                  unsigned short* __restrict__ Ohb,
                                  float* __restrict__ att_map) {
  __shared__ __attribute__((aligned(16))) unsigned short Pl[16 * 32];

  const int lane = threadIdx.x;
  const int qt = blockIdx.x * 16;
  const int h  = blockIdx.y;
  const int hb = h * DH;

  const float scale = 0.125f;          // 1/sqrt(64)
  const float inv_h = 1.0f / (float)NHEAD;

  v16bf a0 = load_a_frag(Qh, qt, DM, hb + 0,  lane);
  v16bf a1 = load_a_frag(Qh, qt, DM, hb + 32, lane);

  v8f o[4];
  float m[8], l[8];
#pragma unroll
  for (int t = 0; t < 4; ++t)
#pragma unroll
    for (int r = 0; r < 8; ++r) o[t][r] = 0.0f;
#pragma unroll
  for (int r = 0; r < 8; ++r) { m[r] = -3.0e38f; l[r] = 0.0f; }

  const int col = lane & 15;
  const int rb  = (lane >> 4) << 3;
  const unsigned short* vhead = Vt + (size_t)h * DH * NK;

  for (int kc = 0; kc < NK; kc += 32) {
    v8f s0 = {0.f, 0.f, 0.f, 0.f, 0.f, 0.f, 0.f, 0.f};
    v8f s1 = {0.f, 0.f, 0.f, 0.f, 0.f, 0.f, 0.f, 0.f};
    {
      v16bf b00 = load_b_frag(Kh, kc,      DM, hb + 0,  lane);
      v16bf b01 = load_b_frag(Kh, kc,      DM, hb + 32, lane);
      s0 = wmma_bf16(a0, b00, s0);
      s0 = wmma_bf16(a1, b01, s0);
      v16bf b10 = load_b_frag(Kh, kc + 16, DM, hb + 0,  lane);
      v16bf b11 = load_b_frag(Kh, kc + 16, DM, hb + 32, lane);
      s1 = wmma_bf16(a0, b10, s1);
      s1 = wmma_bf16(a1, b11, s1);
    }

    float f[8];
#pragma unroll
    for (int r = 0; r < 8; ++r) {
      s0[r] *= scale;
      s1[r] *= scale;
      const int qrow = qt + rb + r;
      atomicAdd(&att_map[(size_t)qrow * NK + kc + col],      s0[r] * inv_h);
      atomicAdd(&att_map[(size_t)qrow * NK + kc + 16 + col], s1[r] * inv_h);

      float cm = fmaxf(s0[r], s1[r]);
      cm = fmaxf(cm, __shfl_xor(cm, 1, 32));
      cm = fmaxf(cm, __shfl_xor(cm, 2, 32));
      cm = fmaxf(cm, __shfl_xor(cm, 4, 32));
      cm = fmaxf(cm, __shfl_xor(cm, 8, 32));
      const float mn = fmaxf(m[r], cm);
      f[r] = __expf(m[r] - mn);
      m[r] = mn;
      const float p0 = __expf(s0[r] - mn);
      const float p1 = __expf(s1[r] - mn);
      float rs = p0 + p1;
      rs += __shfl_xor(rs, 1, 32);
      rs += __shfl_xor(rs, 2, 32);
      rs += __shfl_xor(rs, 4, 32);
      rs += __shfl_xor(rs, 8, 32);
      l[r] = l[r] * f[r] + rs;
      s0[r] = p0;
      s1[r] = p1;
    }

#pragma unroll
    for (int t = 0; t < 4; ++t)
#pragma unroll
      for (int r = 0; r < 8; ++r) o[t][r] *= f[r];

#pragma unroll
    for (int r = 0; r < 8; ++r) {
      Pl[(rb + r) * 32 + col]      = f2bf(s0[r]);
      Pl[(rb + r) * 32 + 16 + col] = f2bf(s1[r]);
    }
    __syncthreads();
    v16bf pa = load_a_frag(Pl, 0, 32, 0, lane);

    {
      v16bf vb0 = load_b_frag(vhead, 0,  NK, kc, lane);
      o[0] = wmma_bf16(pa, vb0, o[0]);
      v16bf vb1 = load_b_frag(vhead, 16, NK, kc, lane);
      o[1] = wmma_bf16(pa, vb1, o[1]);
      v16bf vb2 = load_b_frag(vhead, 32, NK, kc, lane);
      o[2] = wmma_bf16(pa, vb2, o[2]);
      v16bf vb3 = load_b_frag(vhead, 48, NK, kc, lane);
      o[3] = wmma_bf16(pa, vb3, o[3]);
    }
    __syncthreads();
  }

  float inv[8];
#pragma unroll
  for (int r = 0; r < 8; ++r) inv[r] = 1.0f / l[r];
#pragma unroll
  for (int t = 0; t < 4; ++t)
#pragma unroll
    for (int r = 0; r < 8; ++r)
      Ohb[(size_t)(qt + rb + r) * DM + hb + t * 16 + col] = f2bf(o[t][r] * inv[r]);
}

// ---------- launch ----------

extern "C" void kernel_launch(void* const* d_in, const int* in_sizes, int n_in,
                              void* d_out, int out_size, void* d_ws, size_t ws_size,
                              hipStream_t stream) {
  (void)in_sizes; (void)n_in; (void)out_size; (void)ws_size;

  const float* queries = (const float*)d_in[0];
  const float* keys    = (const float*)d_in[1];
  const float* values  = (const float*)d_in[2];
  const float* Wq = (const float*)d_in[3];
  const float* bq = (const float*)d_in[4];
  const float* Wk = (const float*)d_in[5];
  const float* bk = (const float*)d_in[6];
  const float* Wv = (const float*)d_in[7];
  const float* bv = (const float*)d_in[8];
  const float* Wo = (const float*)d_in[9];
  const float* bo = (const float*)d_in[10];

  float* out     = (float*)d_out;
  float* att_map = out + (size_t)NQ * DM;

  const size_t MB = (size_t)1 << 20;
  char* ws = (char*)d_ws;
  unsigned short* qb  = (unsigned short*)(ws + 0 * MB);   // 4 MB
  unsigned short* kb  = (unsigned short*)(ws + 4 * MB);   // 4 MB
  unsigned short* vb_ = (unsigned short*)(ws + 8 * MB);   // 4 MB
  unsigned short* wqb = (unsigned short*)(ws + 12 * MB);  // 2 MB
  unsigned short* wkb = (unsigned short*)(ws + 14 * MB);  // 2 MB
  unsigned short* wvb = (unsigned short*)(ws + 16 * MB);  // 2 MB
  unsigned short* wob = (unsigned short*)(ws + 18 * MB);  // 2 MB
  unsigned short* Qh  = (unsigned short*)(ws + 20 * MB);  // 4 MB
  unsigned short* Kh  = (unsigned short*)(ws + 24 * MB);  // 4 MB
  unsigned short* Vt  = (unsigned short*)(ws + 28 * MB);  // 4 MB
  unsigned short* Ohb = (unsigned short*)(ws + 32 * MB);  // 4 MB  (total 36 MB)

  const int T = 256;
  const int nAct = NQ * DM;       // 2M
  const int nW   = DM * DM;       // 1M

  cvt_bf16_kernel<<<nAct / T, T, 0, stream>>>(queries, qb, nAct);
  cvt_bf16_kernel<<<nAct / T, T, 0, stream>>>(keys,    kb, nAct);
  cvt_bf16_kernel<<<nAct / T, T, 0, stream>>>(values,  vb_, nAct);
  cvt_bf16_kernel<<<nW / T,  T, 0, stream>>>(Wq, wqb, nW);
  cvt_bf16_kernel<<<nW / T,  T, 0, stream>>>(Wk, wkb, nW);
  cvt_bf16_kernel<<<nW / T,  T, 0, stream>>>(Wv, wvb, nW);
  cvt_bf16_kernel<<<nW / T,  T, 0, stream>>>(Wo, wob, nW);

  zero_f32_kernel<<<(NQ * NK) / T, T, 0, stream>>>(att_map, NQ * NK);

  dim3 gGemm(DM / 64, NQ / 128);  // (16, 16), 128 threads = 4 waves
  gemm_bf16_kernel<<<gGemm, 128, 0, stream>>>(qb,  wqb, bq, Qh, NQ, DM, DM, 0);
  gemm_bf16_kernel<<<gGemm, 128, 0, stream>>>(kb,  wkb, bk, Kh, NQ, DM, DM, 0);
  gemm_bf16_kernel<<<gGemm, 128, 0, stream>>>(vb_, wvb, bv, Vt, NQ, DM, DM, 1);

  dim3 gAttn(NQ / 16, NHEAD);     // (128, 16)
  flash_attn_kernel<<<gAttn, 32, 0, stream>>>(Qh, Kh, Vt, Ohb, att_map);

  gemm_bf16_kernel<<<gGemm, 128, 0, stream>>>(Ohb, wob, bo, out, NQ, DM, DM, 2);
}
